// AttentionalPooler_50878182588542
// MI455X (gfx1250) — matrix operand order repaired and checked by reading, not compiled, not run on hardware
//
#include <hip/hip_runtime.h>
#include <hip/hip_bf16.h>

#define DM 1024   // d_model
#define CD 1408   // context dim
#define NH 8      // heads
#define NQh 256   // n_queries
#define BBATCH 16
#define SSEQ 1024
#define HDim 128  // head dim

typedef __attribute__((ext_vector_type(16))) _Float16 v16h;
typedef __attribute__((ext_vector_type(8)))  _Float16 v8h;
typedef __attribute__((ext_vector_type(8)))  float    v8f;

union Frag16 { v16h f; v8h h2[2]; };

__device__ __forceinline__ v8f zero8() {
  v8f z;
  #pragma unroll
  for (int r = 0; r < 8; ++r) z[r] = 0.f;
  return z;
}

// A fragment (16x32 f16, M x K) for v_wmma_f32_16x16x32_f16.
// lanes 0-15 hold K {0..7, 16..23}, lanes 16-31 hold K {8..15, 24..31}; row = lane%16.
__device__ __forceinline__ v16h load_frag_a_g(const _Float16* __restrict__ A, int lda,
                                              int row0, int k0, int lane) {
  Frag16 fr;
  const _Float16* p = A + (size_t)(row0 + (lane & 15)) * lda + k0 + ((lane >> 4) << 3);
  fr.h2[0] = *(const v8h*)(p);        // K = k0 + half*8 .. +7
  fr.h2[1] = *(const v8h*)(p + 16);   // K = k0 + 16 + half*8 .. +7
  return fr.f;
}

// B fragment (32x16 f16, K x N) where B[k][n] = Bsrc[n][k] (NT: column n contiguous in k).
// lane holds col N = lane%16, K = k0 + (lane/16)*8 .. +7 and k0 + 16 + (lane/16)*8 .. +7.
__device__ __forceinline__ v16h load_frag_b_g(const _Float16* __restrict__ Bm, int ldb,
                                              int n0, int k0, int lane) {
  Frag16 fr;
  const _Float16* p = Bm + (size_t)(n0 + (lane & 15)) * ldb + k0 + ((lane >> 4) << 3);
  fr.h2[0] = *(const v8h*)(p);
  fr.h2[1] = *(const v8h*)(p + 16);
  return fr.f;
}

// ---------------- fp32 -> f16 convert ----------------
__global__ __launch_bounds__(256) void cvt_f32_f16(const float* __restrict__ s,
                                                   _Float16* __restrict__ d, int n) {
  int i = blockIdx.x * blockDim.x + threadIdx.x;
  if (i < n) d[i] = (_Float16)s[i];
}

// ---------------- LayerNorm: one wave32 per row ----------------
__global__ __launch_bounds__(256) void ln_rows(const float* __restrict__ X,
                                               const float* __restrict__ w,
                                               const float* __restrict__ bb,
                                               _Float16* __restrict__ out,
                                               int rows, int cols) {
  const int gwv  = (blockIdx.x * blockDim.x + threadIdx.x) >> 5;
  const int lane = threadIdx.x & 31;
  if (gwv >= rows) return;
  const float* x = X + (size_t)gwv * cols;
  float s = 0.f;
  for (int c = lane; c < cols; c += 32) s += x[c];
  #pragma unroll
  for (int o = 16; o > 0; o >>= 1) s += __shfl_xor(s, o);
  const float mean = s * (1.f / (float)cols);
  float v = 0.f;
  for (int c = lane; c < cols; c += 32) { float d = x[c] - mean; v += d * d; }
  #pragma unroll
  for (int o = 16; o > 0; o >>= 1) v += __shfl_xor(v, o);
  const float rstd = rsqrtf(v * (1.f / (float)cols) + 1e-5f);
  _Float16* o_ = out + (size_t)gwv * cols;
  for (int c = lane; c < cols; c += 32)
    o_[c] = (_Float16)((x[c] - mean) * rstd * w[c] + bb[c]);
}

// ---------------- generic NT WMMA GEMM: out = A[M,K] @ Bsrc[N,K]^T + bias -------------
// mode 0/1: f16 row-major out (mode 0 applies scale for Q)
// mode 2  : f16 transposed "Vt" store: out[((b*NH+h)*HDim + d)*SSEQ + s]
// mode 3  : f32 row-major out
__global__ __launch_bounds__(128) void gemm_nt(const _Float16* __restrict__ A,
                                               const _Float16* __restrict__ Bw,
                                               const float* __restrict__ bias,
                                               _Float16* __restrict__ outH,
                                               float* __restrict__ outF,
                                               int M, int N, int K, float scale, int mode) {
  const int lane = threadIdx.x & 31;
  const int gw   = (blockIdx.x << 2) + (threadIdx.x >> 5);   // 4 waves / block
  const int nBlocks = N >> 6;
  const int mTile = gw / nBlocks, nTile = gw % nBlocks;
  const int m0 = mTile << 6, n0 = nTile << 6;
  if (m0 >= M) return;

  v8f acc[4][4];
  #pragma unroll
  for (int i = 0; i < 4; ++i)
    #pragma unroll
    for (int j = 0; j < 4; ++j) acc[i][j] = zero8();

  for (int k0 = 0; k0 < K; k0 += 32) {
    v16h af[4], bf[4];
    #pragma unroll
    for (int i = 0; i < 4; ++i) af[i] = load_frag_a_g(A, K, m0 + (i << 4), k0, lane);
    #pragma unroll
    for (int j = 0; j < 4; ++j) bf[j] = load_frag_b_g(Bw, K, n0 + (j << 4), k0, lane);
    #pragma unroll
    for (int i = 0; i < 4; ++i)
      #pragma unroll
      for (int j = 0; j < 4; ++j)
        acc[i][j] = __builtin_amdgcn_wmma_f32_16x16x32_f16(
            false, af[i], false, bf[j], (short)0, acc[i][j], false, false);
  }

  #pragma unroll
  for (int j = 0; j < 4; ++j) {
    const int n = n0 + (j << 4) + (lane & 15);
    const float bv = bias[n];
    #pragma unroll
    for (int i = 0; i < 4; ++i) {
      const int mr = m0 + (i << 4) + ((lane >> 4) << 3);  // first of 8 consecutive rows
      if (mode == 2) {
        const int hh = n >> 7, d = n & 127;
        const int bb = mr >> 10, s0 = mr & 1023;
        v8h pk;
        #pragma unroll
        for (int r = 0; r < 8; ++r) pk[r] = (_Float16)(acc[i][j][r] + bv);
        *(v8h*)(outH + ((size_t)((bb * NH + hh) * HDim + d) << 10) + s0) = pk;
      } else if (mode == 3) {
        #pragma unroll
        for (int r = 0; r < 8; ++r)
          outF[(size_t)(mr + r) * N + n] = acc[i][j][r] + bv;
      } else {
        #pragma unroll
        for (int r = 0; r < 8; ++r)
          outH[(size_t)(mr + r) * N + n] = (_Float16)((acc[i][j][r] + bv) * scale);
      }
    }
  }
}

// ---------------- attention: one block per (b, h, 64-query tile) ----------------
#define QT 64
__global__ __launch_bounds__(256) void attn_kernel(const _Float16* __restrict__ Qh,
                                                   const _Float16* __restrict__ Kh,
                                                   const _Float16* __restrict__ Vt,
                                                   const unsigned char* __restrict__ mask,
                                                   _Float16* __restrict__ ctx) {
  extern __shared__ float lds[];            // QT*SSEQ scores + QT*4 reduce
  float* red = lds + QT * SSEQ;

  const int b    = blockIdx.x >> 5;
  const int h    = (blockIdx.x >> 2) & 7;
  const int qt   = blockIdx.x & 3;
  const int tid  = threadIdx.x;
  const int lane = tid & 31;
  const int wave = tid >> 5;

  const _Float16* Qb = Qh + (size_t)(qt * QT) * DM + h * HDim;       // lda = DM
  const _Float16* Kb = Kh + (size_t)b * SSEQ * DM + h * HDim;        // rows = tokens, ld = DM
  const _Float16* Vb = Vt + (size_t)(b * NH + h) * HDim * SSEQ;      // [HDim][SSEQ]

  // ---- pass 1: scores = Qtile @ K^T (+ mask), into LDS fp32 ----
  for (int t = wave; t < 4 * (SSEQ / 16); t += 8) {
    const int mi = t & 3, nt = t >> 2;
    v8f acc = zero8();
    #pragma unroll
    for (int kk = 0; kk < 4; ++kk) {
      v16h af = load_frag_a_g(Qb, DM, mi << 4, kk << 5, lane);
      v16h bf = load_frag_b_g(Kb, DM, nt << 4, kk << 5, lane);
      acc = __builtin_amdgcn_wmma_f32_16x16x32_f16(false, af, false, bf, (short)0, acc,
                                                   false, false);
    }
    const int sc = (nt << 4) + (lane & 15);
    const float madd = mask[b * SSEQ + sc] ? 0.f : -3.402823466e38f;
    const int r0 = (mi << 4) + ((lane >> 4) << 3);
    #pragma unroll
    for (int r = 0; r < 8; ++r) lds[(size_t)(r0 + r) * SSEQ + sc] = acc[r] + madd;
  }
  __syncthreads();

  // ---- softmax in LDS: 4 threads per row ----
  const int row = tid >> 2, seg = tid & 3;
  float* prow = lds + (size_t)row * SSEQ + seg * 256;
  float mx = -3.402823466e38f;
  for (int c = 0; c < 256; ++c) mx = fmaxf(mx, prow[c]);
  red[(row << 2) + seg] = mx;
  __syncthreads();
  mx = fmaxf(fmaxf(red[row << 2], red[(row << 2) + 1]),
             fmaxf(red[(row << 2) + 2], red[(row << 2) + 3]));
  __syncthreads();
  float sum = 0.f;
  for (int c = 0; c < 256; ++c) { float e = __expf(prow[c] - mx); prow[c] = e; sum += e; }
  red[(row << 2) + seg] = sum;
  __syncthreads();
  sum = red[row << 2] + red[(row << 2) + 1] + red[(row << 2) + 2] + red[(row << 2) + 3];
  const float inv = 1.f / sum;
  for (int c = 0; c < 256; ++c) prow[c] *= inv;
  __syncthreads();

  // ---- pass 2: ctx = probs(LDS) @ V ; wave owns (mi, 4 n-tiles) ----
  const int mi = wave & 3;
  const int nBase = (wave >> 2) << 2;      // 0 or 4
  v8f acc2[4];
  #pragma unroll
  for (int j = 0; j < 4; ++j) acc2[j] = zero8();

  for (int k0 = 0; k0 < SSEQ; k0 += 32) {
    Frag16 af;
    const float* pa = lds + (size_t)((mi << 4) + (lane & 15)) * SSEQ + k0 + ((lane >> 4) << 3);
    #pragma unroll
    for (int r = 0; r < 8; ++r) af.f[r] = (_Float16)pa[r];
    #pragma unroll
    for (int r = 0; r < 8; ++r) af.f[8 + r] = (_Float16)pa[16 + r];
    #pragma unroll
    for (int j = 0; j < 4; ++j) {
      v16h bf = load_frag_b_g(Vb, SSEQ, (nBase + j) << 4, k0, lane);
      acc2[j] = __builtin_amdgcn_wmma_f32_16x16x32_f16(false, af.f, false, bf, (short)0,
                                                       acc2[j], false, false);
    }
  }
  #pragma unroll
  for (int j = 0; j < 4; ++j) {
    const int d  = h * HDim + ((nBase + j) << 4) + (lane & 15);
    const int q0 = qt * QT + (mi << 4) + ((lane >> 4) << 3);
    #pragma unroll
    for (int r = 0; r < 8; ++r)
      ctx[(size_t)(b * NQh + q0 + r) * DM + d] = (_Float16)acc2[j][r];
  }
}

extern "C" void kernel_launch(void* const* d_in, const int* in_sizes, int n_in,
                              void* d_out, int out_size, void* d_ws, size_t ws_size,
                              hipStream_t stream) {
  const float* x            = (const float*)d_in[0];
  const unsigned char* mask = (const unsigned char*)d_in[1];
  const float* query        = (const float*)d_in[2];
  const float* ln_q_w       = (const float*)d_in[3];
  const float* ln_q_b       = (const float*)d_in[4];
  const float* ln_k_w       = (const float*)d_in[5];
  const float* ln_k_b       = (const float*)d_in[6];
  const float* Wq = (const float*)d_in[7];
  const float* Wk = (const float*)d_in[8];
  const float* Wv = (const float*)d_in[9];
  const float* bq = (const float*)d_in[10];
  const float* bk = (const float*)d_in[11];
  const float* bv = (const float*)d_in[12];
  const float* Wo = (const float*)d_in[13];
  const float* bo = (const float*)d_in[14];
  float* out = (float*)d_out;

  _Float16* wsH = (_Float16*)d_ws;
  const size_t OFF_QLN = 0;                                   // 256*1024
  const size_t OFF_KVH = OFF_QLN + (size_t)NQh * DM;          // 16384*1408
  const size_t OFF_WQH = OFF_KVH + (size_t)BBATCH * SSEQ * CD;
  const size_t OFF_WKH = OFF_WQH + (size_t)DM * DM;
  const size_t OFF_WVH = OFF_WKH + (size_t)DM * CD;
  const size_t OFF_WOH = OFF_WVH + (size_t)DM * CD;
  const size_t OFF_QH  = OFF_WOH + (size_t)DM * DM;
  const size_t OFF_KH  = OFF_QH  + (size_t)NQh * DM;
  const size_t OFF_VT  = OFF_KH  + (size_t)BBATCH * SSEQ * DM;
  const size_t OFF_CTX = OFF_VT  + (size_t)BBATCH * SSEQ * DM;

  _Float16* qln  = wsH + OFF_QLN;
  _Float16* kvh  = wsH + OFF_KVH;
  _Float16* Wqh  = wsH + OFF_WQH;
  _Float16* Wkh  = wsH + OFF_WKH;
  _Float16* Wvh  = wsH + OFF_WVH;
  _Float16* Woh  = wsH + OFF_WOH;
  _Float16* Qh   = wsH + OFF_QH;
  _Float16* Kh   = wsH + OFF_KH;
  _Float16* Vtp  = wsH + OFF_VT;
  _Float16* ctxh = wsH + OFF_CTX;

  // weight conversions fp32 -> f16
  cvt_f32_f16<<<(DM * DM + 255) / 256, 256, 0, stream>>>(Wq, Wqh, DM * DM);
  cvt_f32_f16<<<(DM * CD + 255) / 256, 256, 0, stream>>>(Wk, Wkh, DM * CD);
  cvt_f32_f16<<<(DM * CD + 255) / 256, 256, 0, stream>>>(Wv, Wvh, DM * CD);
  cvt_f32_f16<<<(DM * DM + 255) / 256, 256, 0, stream>>>(Wo, Woh, DM * DM);

  // layernorms (wave per row)
  ln_rows<<<(NQh * 32) / 256, 256, 0, stream>>>(query, ln_q_w, ln_q_b, qln, NQh, DM);
  ln_rows<<<(BBATCH * SSEQ * 32) / 256, 256, 0, stream>>>(x, ln_k_w, ln_k_b, kvh,
                                                          BBATCH * SSEQ, CD);

  const float qscale = 0.08838834764831845f;  // 128^-0.5

  // Q = (ln_q @ Wq^T + bq) * scale       [256 x 1024]
  gemm_nt<<<((NQh / 64) * (DM / 64)) / 4, 128, 0, stream>>>(qln, Wqh, bq, Qh, nullptr,
                                                            NQh, DM, DM, qscale, 0);
  // K = kv @ Wk^T + bk                   [16384 x 1024] row-major
  gemm_nt<<<((BBATCH * SSEQ / 64) * (DM / 64)) / 4, 128, 0, stream>>>(
      kvh, Wkh, bk, Kh, nullptr, BBATCH * SSEQ, DM, CD, 1.f, 1);
  // V = kv @ Wv^T + bv                   stored transposed per (b,h): [B][H][128][S]
  gemm_nt<<<((BBATCH * SSEQ / 64) * (DM / 64)) / 4, 128, 0, stream>>>(
      kvh, Wvh, bv, Vtp, nullptr, BBATCH * SSEQ, DM, CD, 1.f, 2);

  // attention: scores->softmax->ctx; 256 KB + 1 KB dynamic LDS (WGP has 320 KB)
  attn_kernel<<<BBATCH * NH * (NQh / QT), 256, (QT * SSEQ + QT * 4) * sizeof(float),
                stream>>>(Qh, Kh, Vtp, mask, ctxh);

  // out = ctx @ Wo^T + bo  (fp32)        [4096 x 1024]
  gemm_nt<<<((BBATCH * NQh / 64) * (DM / 64)) / 4, 128, 0, stream>>>(
      ctxh, Woh, bo, nullptr, out, BBATCH * NQh, DM, DM, 1.f, 3);
}
